// _LlamaAttention_29635274342443
// MI455X (gfx1250) — matrix operand order repaired and checked by reading, not compiled, not run on hardware
//
#include <hip/hip_runtime.h>
#include <hip/hip_bf16.h>
#include <math.h>

// ---------------------------------------------------------------------------
// Llama attention block for MI455X (gfx1250), wave32 + WMMA bf16 + async LDS.
//   B=2, S=2048, H=4096, NH=32, D=128
// All GEMMs run on v_wmma_f32_16x16x32_bf16 with fp32 accumulation; global ->
// LDS movement uses global_load_async_to_lds_b128 (ASYNCcnt) double-buffered.
// ---------------------------------------------------------------------------

typedef __bf16 bf16;
typedef __attribute__((ext_vector_type(16))) __bf16 v16bf;
typedef __attribute__((ext_vector_type(8)))  __bf16 v8bf;
typedef __attribute__((ext_vector_type(8)))  float  v8f;

#define B_   2
#define S_   2048
#define H_   4096
#define NH_  32
#define D_   128
#define BS_  (B_ * S_)      // 4096
#define KDIM 4096           // every projection GEMM is 4096x4096x4096

// ---------------------------------------------------------------------------
// helpers
// ---------------------------------------------------------------------------
__device__ __forceinline__ v8bf load8(const bf16* p) {
    return *reinterpret_cast<const v8bf*>(p);
}
__device__ __forceinline__ v16bf load16(const bf16* p) {
    return *reinterpret_cast<const v16bf*>(p);
}
__device__ __forceinline__ v16bf cat16(v8bf lo, v8bf hi) {
    v16bf r;
#pragma unroll
    for (int i = 0; i < 8; ++i) { r[i] = lo[i]; r[i + 8] = hi[i]; }
    return r;
}
__device__ __forceinline__ v8f vzero8() {
    v8f r;
#pragma unroll
    for (int i = 0; i < 8; ++i) r[i] = 0.0f;
    return r;
}
__device__ __forceinline__ v8f wmma_bf16(v16bf a, v16bf b, v8f c) {
    return __builtin_amdgcn_wmma_f32_16x16x32_bf16(
        /*neg_a=*/false, a, /*neg_b=*/false, b,
        /*c_mod=*/(short)0, c, /*reuse_a=*/false, /*reuse_b=*/false);
}

// async 16B copy global -> LDS (ASYNCcnt-tracked CDNA5 path).
// vdst VGPR carries the wave-relative LDS byte offset, vaddr the 64b address.
__device__ __forceinline__ void async_cp16(unsigned int lds_off,
                                           unsigned long long gaddr) {
    asm volatile("global_load_async_to_lds_b128 %0, %1, off"
                 :: "v"(lds_off), "v"(gaddr) : "memory");
}
__device__ __forceinline__ void wait_async0() {
    asm volatile("s_wait_asynccnt 0" ::: "memory");
}
__device__ __forceinline__ unsigned int lds_off(const void* p) {
    // generic pointers to LDS keep the allocation-relative offset in [31:0]
    return (unsigned int)(uintptr_t)p;
}

// ---------------------------------------------------------------------------
// Stage 0: fp32 -> bf16 conversion (all operands then live in the 192MB L2)
// ---------------------------------------------------------------------------
__global__ void cvt_f32_bf16(const float* __restrict__ in,
                             bf16* __restrict__ out, int n) {
    int i = blockIdx.x * blockDim.x + threadIdx.x;
    if (i < n) out[i] = (bf16)in[i];
}

// ---------------------------------------------------------------------------
// Stage 1/3/5: C = A * W^T  (einsum 'bsh,oh->bso' is exactly A * W^T)
// Block = 256 threads = 8 waves (4M x 2N); block tile 128x128, wave tile
// 32x64 (2x4 WMMA accumulators); K in steps of 32.
// A/B 128x32 bf16 tiles are DMA'd into double-buffered LDS with
// global_load_async_to_lds_b128; fragments are re-read as ds_load_b128 with a
// 16B XOR swizzle (chunk ^= (row>>2)&3) to spread LDS banks.
//   MODE 0: out = bf16 [M][N]                (Q_raw / K_raw pre-RoPE)
//   MODE 1: out = bf16 V^T [B][NH][D][S]     (transposed for P*V fragments)
//   MODE 2: out = f32  [M][N]                (final output projection)
// ---------------------------------------------------------------------------
__device__ __forceinline__ void gemm_issue_tile(
    const char* Abytes, const char* Wbytes, unsigned int abase,
    unsigned int bbase, int k0, int tid) {
#pragma unroll
    for (int i = 0; i < 2; ++i) {
        const int c   = tid + i * 256;   // 0..511 : 16B chunks of a 8KB tile
        const int row = c >> 2;          // 64B (4 chunks) per row of 32 bf16
        const int ck  = c & 3;
        const int sw  = (row >> 2) & 3;  // bank swizzle
        const unsigned int loff = row * 64 + ((ck ^ sw) << 4);
        async_cp16(abase + loff,
                   (unsigned long long)Abytes + (size_t)row * (KDIM * 2) +
                       (size_t)k0 * 2 + (ck << 4));
        async_cp16(bbase + loff,
                   (unsigned long long)Wbytes + (size_t)row * (KDIM * 2) +
                       (size_t)k0 * 2 + (ck << 4));
    }
}

template <int MODE>
__global__ __launch_bounds__(256) void gemm_nt(const bf16* __restrict__ A,
                                               const bf16* __restrict__ W,
                                               void* __restrict__ outp) {
    __shared__ bf16 At[2][128 * 32];   // [row][k], 16B-chunk swizzled
    __shared__ bf16 Bt[2][128 * 32];   // [col][k], 16B-chunk swizzled

    const int tid  = threadIdx.x;
    const int lane = tid & 31;
    const int wv   = tid >> 5;
    const int wm   = wv & 3;           // 4 waves along M
    const int wn   = wv >> 2;          // 2 waves along N
    const int half = lane >> 4;
    const int l16  = lane & 15;
    const int m0   = blockIdx.y * 128;
    const int n0   = blockIdx.x * 128;

    const unsigned int ab[2] = {lds_off(&At[0][0]), lds_off(&At[1][0])};
    const unsigned int bb[2] = {lds_off(&Bt[0][0]), lds_off(&Bt[1][0])};
    const char* Abytes = (const char*)(A + (size_t)m0 * KDIM);
    const char* Wbytes = (const char*)(W + (size_t)n0 * KDIM);

    v8f acc[2][4];
#pragma unroll
    for (int tm = 0; tm < 2; ++tm)
#pragma unroll
        for (int tn = 0; tn < 4; ++tn) acc[tm][tn] = vzero8();

    gemm_issue_tile(Abytes, Wbytes, ab[0], bb[0], 0, tid);

    const int NKT = KDIM / 32;
    for (int kt = 0; kt < NKT; ++kt) {
        wait_async0();       // this wave's DMA for tile kt has landed in LDS
        __syncthreads();     // everyone's has; also fences buf reuse
        if (kt + 1 < NKT)    // overlap next tile's DMA with this tile's WMMAs
            gemm_issue_tile(Abytes, Wbytes, ab[(kt + 1) & 1], bb[(kt + 1) & 1],
                            (kt + 1) * 32, tid);

        const bf16* Ab_ = &At[kt & 1][0];
        const bf16* Bb_ = &Bt[kt & 1][0];

        // A fragments (16x32): row = l16, 16 contiguous K at 16*half
        v16bf af[2];
#pragma unroll
        for (int tm = 0; tm < 2; ++tm) {
            const int r  = wm * 32 + tm * 16 + l16;
            const int sw = (r >> 2) & 3;
            af[tm] = cat16(load8(Ab_ + r * 32 + (((2 * half) ^ sw) << 3)),
                           load8(Ab_ + r * 32 + (((2 * half + 1) ^ sw) << 3)));
        }
        // B fragments (32x16): col = l16, 16 contiguous K at 16*half
#pragma unroll
        for (int tn = 0; tn < 4; ++tn) {
            const int cN = wn * 64 + tn * 16 + l16;
            const int sw = (cN >> 2) & 3;
            v16bf bfr = cat16(
                load8(Bb_ + cN * 32 + (((2 * half) ^ sw) << 3)),
                load8(Bb_ + cN * 32 + (((2 * half + 1) ^ sw) << 3)));
#pragma unroll
            for (int tm = 0; tm < 2; ++tm)
                acc[tm][tn] = wmma_bf16(af[tm], bfr, acc[tm][tn]);
        }
        // NOTE: no tail barrier needed — reaching the next iteration's
        // __syncthreads() implies every wave finished this compute before
        // anyone overwrites this buffer (issued two iterations later).
    }

    // C/D layout: VGPR j -> row j + 8*half, col = l16
#pragma unroll
    for (int tm = 0; tm < 2; ++tm)
#pragma unroll
        for (int tn = 0; tn < 4; ++tn)
#pragma unroll
            for (int j = 0; j < 8; ++j) {
                const int row = m0 + wm * 32 + tm * 16 + j + 8 * half;
                const int col = n0 + wn * 64 + tn * 16 + l16;
                const float val = acc[tm][tn][j];
                if (MODE == 2) {
                    ((float*)outp)[(size_t)row * H_ + col] = val;
                } else if (MODE == 0) {
                    ((bf16*)outp)[(size_t)row * H_ + col] = (bf16)val;
                } else {  // MODE == 1 : V^T [B][NH][D][S]
                    const int b = row >> 11, s = row & (S_ - 1);
                    const int h = col >> 7,  d = col & (D_ - 1);
                    ((bf16*)outp)[(((size_t)(b * NH_ + h)) * D_ + d) * S_ + s] =
                        (bf16)val;
                }
            }
}

// ---------------------------------------------------------------------------
// Stage 2: RoPE on Q and K, re-layout to [B][NH][S][D] bf16.
// rope_table row = [sin(0..63) | cos(0..63)]
// ---------------------------------------------------------------------------
__global__ __launch_bounds__(256) void rope_kernel(
    const bf16* __restrict__ Qraw, const bf16* __restrict__ Kraw,
    const int* __restrict__ pos_ids, const float* __restrict__ rope,
    bf16* __restrict__ Qo, bf16* __restrict__ Ko) {
    int idx = blockIdx.x * blockDim.x + threadIdx.x;
    if (idx >= B_ * S_ * NH_ * (D_ / 2)) return;
    const int d = idx & 63; int t = idx >> 6;
    const int h = t & (NH_ - 1); t >>= 5;
    const int s = t & (S_ - 1);  const int b = t >> 11;

    const int   pos = pos_ids[b * S_ + s];
    const float sn  = rope[(size_t)pos * D_ + d];
    const float cs  = rope[(size_t)pos * D_ + 64 + d];

    const size_t rbase = ((size_t)(b * S_ + s)) * H_ + (size_t)h * D_;
    const size_t obase = (((size_t)b * NH_ + h) * S_ + s) * (size_t)D_;
    {
        const float x1 = (float)Qraw[rbase + d];
        const float x2 = (float)Qraw[rbase + 64 + d];
        Qo[obase + d]      = (bf16)(x1 * cs - x2 * sn);
        Qo[obase + 64 + d] = (bf16)(x2 * cs + x1 * sn);
    }
    {
        const float x1 = (float)Kraw[rbase + d];
        const float x2 = (float)Kraw[rbase + 64 + d];
        Ko[obase + d]      = (bf16)(x1 * cs - x2 * sn);
        Ko[obase + 64 + d] = (bf16)(x2 * cs + x1 * sn);
    }
}

// ---------------------------------------------------------------------------
// Stage 4: causal flash attention.
// Block = 8 waves x 16 q rows = 128 q rows, one (b, head).
// K (32x128) and V^T (128x32) tiles are shared block-wide in double-buffered
// LDS via async DMA (eliminates the 8x per-wave reload of round 1); pipeline:
//   wait_async -> barrier -> issue tile kt+1 -> WMMA on tile kt.
// All waves run the block-uniform causal tile count; waves past their causal
// limit skip compute but keep barriers. Per-wave LDS P-tile relayout
// (C-layout -> A-layout) as before, guarded by s_wait_dscnt.
// ---------------------------------------------------------------------------
__device__ __forceinline__ void attn_issue_tile(
    const char* Khb, const char* Vhb, unsigned int kbase, unsigned int vbase,
    int k0, int tid) {
#pragma unroll
    for (int i = 0; i < 2; ++i) {
        const int c = tid + i * 256;     // 0..511 chunks of 16B
        {   // K tile: 32 keys x 256B; swizzle chunk ^= key&15
            const int key = c >> 4, ck = c & 15;
            async_cp16(kbase + key * 256 + ((ck ^ (key & 15)) << 4),
                       (unsigned long long)Khb +
                           (size_t)(k0 + key) * (D_ * 2) + (ck << 4));
        }
        {   // V^T tile: 128 d x 64B; swizzle chunk ^= (d>>2)&3
            const int d = c >> 2, cv = c & 3;
            async_cp16(vbase + d * 64 + ((cv ^ ((d >> 2) & 3)) << 4),
                       (unsigned long long)Vhb +
                           ((size_t)d * S_ + k0) * 2 + (cv << 4));
        }
    }
}

__global__ __launch_bounds__(256) void flash_attn(
    const bf16* __restrict__ Q,   // [B][NH][S][D]
    const bf16* __restrict__ K,   // [B][NH][S][D]
    const bf16* __restrict__ Vt,  // [B][NH][D][S]
    bf16* __restrict__ O) {       // [B*S][H]
    __shared__ bf16 Ktile[2][32 * 128];   // [key][d]   swizzled
    __shared__ bf16 Vtile[2][128 * 32];   // [d][key]   swizzled
    __shared__ bf16 lds_p[8][16 * 32];    // per-wave P tile

    const int tid  = threadIdx.x;
    const int lane = tid & 31;
    const int wv   = tid >> 5;
    const int half = lane >> 4;
    const int l16  = lane & 15;
    const int h    = blockIdx.y;
    const int b    = blockIdx.z;
    const int q0   = blockIdx.x * 128 + wv * 16;  // this wave's first q row

    const bf16* Qh = Q  + (((size_t)b * NH_ + h) * S_) * D_;
    const char* Khb = (const char*)(K  + (((size_t)b * NH_ + h) * S_) * D_);
    const char* Vhb = (const char*)(Vt + (((size_t)b * NH_ + h) * D_) * S_);
    const unsigned int kb[2] = {lds_off(&Ktile[0][0]), lds_off(&Ktile[1][0])};
    const unsigned int vb[2] = {lds_off(&Vtile[0][0]), lds_off(&Vtile[1][0])};

    // Q fragments: 4 x (16x32) covering D=128, loaded once from global/L2
    v16bf qa[4];
#pragma unroll
    for (int t = 0; t < 4; ++t) {
        const bf16* qr = Qh + (size_t)(q0 + l16) * D_ + 32 * t;
        qa[t] = cat16(load8(qr + 16 * half), load8(qr + 16 * half + 8));
    }

    v8f   oacc[8];
    float mrow[8], lrow[8];
#pragma unroll
    for (int n = 0; n < 8; ++n) oacc[n] = vzero8();
#pragma unroll
    for (int j = 0; j < 8; ++j) { mrow[j] = -__builtin_inff(); lrow[j] = 0.f; }

    const float scale = 0.08838834764831845f;  // 1/sqrt(D)
    const int   nkt   = blockIdx.x * 4 + 4;    // block-uniform causal tiles

    attn_issue_tile(Khb, Vhb, kb[0], vb[0], 0, tid);

    for (int kt = 0; kt < nkt; ++kt) {
        const int k0 = kt * 32;
        wait_async0();
        __syncthreads();
        if (kt + 1 < nkt)
            attn_issue_tile(Khb, Vhb, kb[(kt + 1) & 1], vb[(kt + 1) & 1],
                            (kt + 1) * 32, tid);

        if (k0 <= q0 + 15) {  // causal skip (barrier structure preserved)
            const bf16* Kt = &Ktile[kt & 1][0];
            const bf16* Vb = &Vtile[kt & 1][0];

            // ---- S = Q * K^T : two 16x16 f32 score tiles
            v8f s0 = vzero8(), s1 = vzero8();
#pragma unroll
            for (int t = 0; t < 4; ++t) {
                const int c0 = 4 * t + 2 * half;
                v16bf kf0 = cat16(
                    load8(Kt + l16 * 128 + ((c0 ^ l16) << 3)),
                    load8(Kt + l16 * 128 + (((c0 + 1) ^ l16) << 3)));
                const int k1 = 16 + l16;
                v16bf kf1 = cat16(
                    load8(Kt + k1 * 128 + ((c0 ^ (k1 & 15)) << 3)),
                    load8(Kt + k1 * 128 + (((c0 + 1) ^ (k1 & 15)) << 3)));
                s0 = wmma_bf16(qa[t], kf0, s0);
                s1 = wmma_bf16(qa[t], kf1, s1);
            }

            // ---- online softmax (rows live on 16-lane halves)
            const int key0 = k0 + l16, key1 = k0 + 16 + l16;
#pragma unroll
            for (int j = 0; j < 8; ++j) {
                const int qrow = q0 + j + 8 * half;
                float v0 = (key0 <= qrow) ? s0[j] * scale : -3.0e38f;
                float v1 = (key1 <= qrow) ? s1[j] * scale : -3.0e38f;

                float mx = fmaxf(v0, v1);
#pragma unroll
                for (int m = 1; m < 16; m <<= 1)
                    mx = fmaxf(mx, __shfl_xor(mx, m, 32));
                const float mnew = fmaxf(mrow[j], mx);
                const float corr = __expf(mrow[j] - mnew);
                const float p0   = __expf(v0 - mnew);
                const float p1   = __expf(v1 - mnew);
                float rs = p0 + p1;
#pragma unroll
                for (int m = 1; m < 16; m <<= 1) rs += __shfl_xor(rs, m, 32);
                lrow[j] = lrow[j] * corr + rs;
                mrow[j] = mnew;
#pragma unroll
                for (int n = 0; n < 8; ++n) oacc[n][j] *= corr;

                const int rloc = j + 8 * half;
                lds_p[wv][rloc * 32 + l16]      = (bf16)p0;
                lds_p[wv][rloc * 32 + 16 + l16] = (bf16)p1;
            }

            // wave-local LDS RAW: order + drain DS counter (no block barrier)
            __builtin_amdgcn_wave_barrier();
            asm volatile("s_wait_dscnt 0" ::: "memory");

            const bf16* pb = &lds_p[wv][l16 * 32];
            v16bf pa = cat16(load8(pb + 16 * half), load8(pb + 16 * half + 8));

            // ---- O += P * V from swizzled V^T LDS tile
#pragma unroll
            for (int n = 0; n < 8; ++n) {
                const int d  = n * 16 + l16;
                const int sw = (d >> 2) & 3;
                v16bf vf = cat16(
                    load8(Vb + d * 32 + (((2 * half) ^ sw) << 3)),
                    load8(Vb + d * 32 + (((2 * half + 1) ^ sw) << 3)));
                oacc[n] = wmma_bf16(pa, vf, oacc[n]);
            }
        }
    }

    // ---- normalize and write attn output as bf16 [B*S][H]
#pragma unroll
    for (int j = 0; j < 8; ++j) {
        const int   qrow = q0 + j + 8 * half;
        const float inv  = 1.0f / lrow[j];
        const size_t base = ((size_t)b * S_ + qrow) * H_ + (size_t)h * D_;
#pragma unroll
        for (int n = 0; n < 8; ++n)
            O[base + n * 16 + l16] = (bf16)(oacc[n][j] * inv);
    }
}

// ---------------------------------------------------------------------------
// launch
// ---------------------------------------------------------------------------
extern "C" void kernel_launch(void* const* d_in, const int* in_sizes, int n_in,
                              void* d_out, int out_size, void* d_ws,
                              size_t ws_size, hipStream_t stream) {
    const float* hidden  = (const float*)d_in[0];
    const int*   pos_ids = (const int*)d_in[1];
    // d_in[2] = attention_mask (causal; applied analytically in flash_attn)
    const float* Wq = (const float*)d_in[3];
    const float* Wk = (const float*)d_in[4];
    const float* Wv = (const float*)d_in[5];
    const float* Wo = (const float*)d_in[6];
    const float* rope = (const float*)d_in[7];

    const size_t MAT = (size_t)4096 * 4096;  // every matrix here is 4096x4096
    bf16* hXb  = (bf16*)d_ws;
    bf16* Wqb  = hXb + MAT;
    bf16* Wkb  = Wqb + MAT;
    bf16* Wvb  = Wkb + MAT;
    bf16* Wob  = Wvb + MAT;
    bf16* Qraw = Wob + MAT;
    bf16* Kraw = Qraw + MAT;
    bf16* Qb   = Kraw + MAT;
    bf16* Kb   = Qb + MAT;
    bf16* Vt   = Kb + MAT;
    bf16* Ab   = Vt + MAT;   // total: 11 * 32MB = 352MB of workspace

    const int n = (int)MAT;
    const dim3 cb(256), cg((n + 255) / 256);
    cvt_f32_bf16<<<cg, cb, 0, stream>>>(hidden, hXb, n);
    cvt_f32_bf16<<<cg, cb, 0, stream>>>(Wq, Wqb, n);
    cvt_f32_bf16<<<cg, cb, 0, stream>>>(Wk, Wkb, n);
    cvt_f32_bf16<<<cg, cb, 0, stream>>>(Wv, Wvb, n);
    cvt_f32_bf16<<<cg, cb, 0, stream>>>(Wo, Wob, n);

    const dim3 gb(256), gg(H_ / 128, BS_ / 128);  // (32, 32)
    gemm_nt<0><<<gg, gb, 0, stream>>>(hXb, Wqb, (void*)Qraw);
    gemm_nt<0><<<gg, gb, 0, stream>>>(hXb, Wkb, (void*)Kraw);
    gemm_nt<1><<<gg, gb, 0, stream>>>(hXb, Wvb, (void*)Vt);

    const int rtot = B_ * S_ * NH_ * (D_ / 2);
    rope_kernel<<<(rtot + 255) / 256, 256, 0, stream>>>(Qraw, Kraw, pos_ids,
                                                        rope, Qb, Kb);

    const dim3 fg(S_ / 128, NH_, B_);
    flash_attn<<<fg, 256, 0, stream>>>(Qb, Kb, Vt, Ab);

    gemm_nt<2><<<gg, gb, 0, stream>>>(Ab, Wob, d_out);
}
